// AttentionBlock_87471303950692
// MI455X (gfx1250) — matrix operand order repaired and checked by reading, not compiled, not run on hardware
//
#include <hip/hip_runtime.h>
#include <hip/hip_bf16.h>

typedef __attribute__((ext_vector_type(16))) _Float16 v16h;
typedef __attribute__((ext_vector_type(8)))  _Float16 v8h;
typedef __attribute__((ext_vector_type(8)))  float    v8f;
typedef int v4i __attribute__((vector_size(16)));

#define BATCH   16
#define CDIM    512
#define NDIM    1024   // 32*32
#define HEADS   8
#define HD      64
#define GROUPS  32
#define CPG     16
#define EPSV    1e-5f

#define QKV_W_ELEMS  (3 * CDIM * CDIM)   // 786432
#define PROJ_W_ELEMS (CDIM * CDIM)       // 262144
#define PLANE ((size_t)BATCH * CDIM * NDIM)

// ---- gfx1250 async global->LDS (ASYNCcnt-tracked) with safe fallback ----
#if __has_builtin(__builtin_amdgcn_global_load_async_to_lds_b128)
#define HAVE_ASYNC_LDS 1
__device__ inline void async_copy_b128(const _Float16* g, _Float16* l) {
  __builtin_amdgcn_global_load_async_to_lds_b128(
      (__attribute__((address_space(1))) v4i*)(g),
      (__attribute__((address_space(3))) v4i*)(l), 0, 0);
}
#endif

// ---------------- WMMA fragment helpers (ISA 7.12.2 layouts) ----------------

// A-matrix 16x32 f16 from row-major [M x K], row stride ld.
// lanes 0-15: M=lane, K={0..7,16..23}; lanes 16-31: M=lane-16, K={8..15,24..31}
__device__ inline v16h load_a_frag(const _Float16* src, int ld) {
  const int lane = threadIdx.x & 31;
  const int m    = lane & 15;
  const int kb   = (lane >> 4) << 3;          // 0 or 8
  const _Float16* p = src + (size_t)m * ld;
  v8h lo = *(const v8h*)(p + kb);
  v8h hi = *(const v8h*)(p + 16 + kb);
  v16h a;
#pragma unroll
  for (int i = 0; i < 8; ++i) { a[i] = lo[i]; a[8 + i] = hi[i]; }
  return a;
}

// B-matrix 32x16 f16 from a K-contiguous (column-major-by-K) source:
// src[(col)*ld + k].  lane = col; lanes 0-15 hold K={0..7,16..23},
// lanes 16-31 hold K={8..15,24..31}.  -> two B128 loads per lane.
__device__ inline v16h load_b_frag_t(const _Float16* src, int ld) {
  const int lane = threadIdx.x & 31;
  const int n    = lane & 15;
  const int kb   = (lane >> 4) << 3;          // 0 or 8
  const _Float16* p = src + (size_t)n * ld + kb;
  v8h lo = *(const v8h*)(p);
  v8h hi = *(const v8h*)(p + 16);
  v16h b;
#pragma unroll
  for (int i = 0; i < 8; ++i) { b[i] = lo[i]; b[8 + i] = hi[i]; }
  return b;
}

__device__ inline v8f wmma_f16(v16h a, v16h b, v8f c) {
  return __builtin_amdgcn_wmma_f32_16x16x32_f16(false, a, false, b, (short)0, c,
                                                false, false);
}

__device__ inline float rmax16(float v) {
#pragma unroll
  for (int m = 1; m < 16; m <<= 1) v = fmaxf(v, __shfl_xor(v, m, 32));
  return v;
}
__device__ inline float rsum16(float v) {
#pragma unroll
  for (int m = 1; m < 16; m <<= 1) v += __shfl_xor(v, m, 32);
  return v;
}

// ---------------- Kernel 0: weights -> f16 ----------------
__global__ void k_cvt(const float* __restrict__ qkv_w,
                      const float* __restrict__ proj_w,
                      _Float16* __restrict__ whq,
                      _Float16* __restrict__ whp) {
  const int i = blockIdx.x * 256 + threadIdx.x;
  if (i < QKV_W_ELEMS) whq[i] = (_Float16)qkv_w[i];
  if (i < PROJ_W_ELEMS) whp[i] = (_Float16)proj_w[i];
}

// ---------------- Kernel 1: GroupNorm -> f16 xn^T [B,N,C] ----------------
__global__ void k_groupnorm(const float* __restrict__ x,
                            const float* __restrict__ gamma,
                            const float* __restrict__ beta,
                            _Float16* __restrict__ xnT) {
  __shared__ float s_sum[256];
  __shared__ float s_ss[256];
  const int bg = blockIdx.x;
  const int b  = bg / GROUPS;
  const int g  = bg % GROUPS;
  const int t  = threadIdx.x;
  const size_t base = ((size_t)b * CDIM + (size_t)g * CPG) * NDIM;
  const int nelem = CPG * NDIM;         // 16384

  float s = 0.f, ss = 0.f;
  for (int i = t; i < nelem; i += 256) {
    float v = x[base + i];
    s += v; ss += v * v;
  }
  s_sum[t] = s; s_ss[t] = ss;
  __syncthreads();
  for (int off = 128; off > 0; off >>= 1) {
    if (t < off) { s_sum[t] += s_sum[t + off]; s_ss[t] += s_ss[t + off]; }
    __syncthreads();
  }
  const float mean = s_sum[0] * (1.0f / nelem);
  const float var  = s_ss[0] * (1.0f / nelem) - mean * mean;
  const float inv  = rsqrtf(var + EPSV);

  for (int i = t; i < nelem; i += 256) {
    const int cl = i >> 10;             // channel within group
    const int n  = i & 1023;
    const int c  = g * CPG + cl;
    float v = (x[base + i] - mean) * inv * gamma[c] + beta[c];
    xnT[((size_t)b * NDIM + n) * CDIM + c] = (_Float16)v;
  }
}

// ---------------- Kernel 2: QKV GEMM (wave = 16 x 64 tile) ----------------
__global__ void k_qkv(const _Float16* __restrict__ xnT,   // [B,N,C]
                      const _Float16* __restrict__ whq,   // [3C,C] f16
                      const float* __restrict__ qkv_b,    // [3C]
                      _Float16* __restrict__ wq,          // [B,H,N,HD]
                      _Float16* __restrict__ wk,          // [B,H,N,HD]
                      _Float16* __restrict__ wv) {        // [B,H,HD,N]
  const int wid  = threadIdx.x >> 5;
  const int lane = threadIdx.x & 31;
  const int gw   = blockIdx.x * 8 + wid;
  const int wavesPerBatch = (3 * CDIM / 16) * (NDIM / 64);  // 96*16
  const int b  = gw / wavesPerBatch;
  const int r  = gw % wavesPerBatch;
  const int o0 = (r / 16) * 16;
  const int n0 = (r % 16) * 64;

  const _Float16* xb = xnT + ((size_t)b * NDIM + n0) * CDIM;
  v8f acc[4] = {};
  for (int kk = 0; kk < CDIM; kk += 32) {
    const v16h a = load_a_frag(whq + (size_t)o0 * CDIM + kk, CDIM);
#pragma unroll
    for (int t = 0; t < 4; ++t) {
      const v16h bm = load_b_frag_t(xb + (size_t)(t * 16) * CDIM + kk, CDIM);
      acc[t] = wmma_f16(a, bm, acc[t]);
    }
  }

  const int half = lane >> 4;
#pragma unroll
  for (int t = 0; t < 4; ++t) {
    const int n = n0 + t * 16 + (lane & 15);
#pragma unroll
    for (int rr = 0; rr < 8; ++rr) {
      const int o = o0 + rr + half * 8;
      float val = acc[t][rr] + qkv_b[o];
      if (o < CDIM) {                       // Q -> [b,h,n,c]
        const int h = o >> 6, c = o & 63;
        wq[(((size_t)b * HEADS + h) * NDIM + n) * HD + c] = (_Float16)val;
      } else if (o < 2 * CDIM) {            // K -> [b,h,n,c]
        const int oo = o - CDIM, h = oo >> 6, c = oo & 63;
        wk[(((size_t)b * HEADS + h) * NDIM + n) * HD + c] = (_Float16)val;
      } else {                              // V -> [b,h,c,n]
        const int oo = o - 2 * CDIM, h = oo >> 6, c = oo & 63;
        wv[(((size_t)b * HEADS + h) * HD + c) * NDIM + n] = (_Float16)val;
      }
    }
  }
}

// ---------------- Kernel 3: flash attention ----------------
// Block = 8 waves, all in the same (b,h); wave owns 16 query rows.
// K/V panels staged cooperatively in LDS (async global->LDS when available).
__global__ void k_attn(const _Float16* __restrict__ wq,  // [B,H,N,HD]
                       const _Float16* __restrict__ wk,  // [B,H,N,HD]
                       const _Float16* __restrict__ wv,  // [B,H,HD,N]
                       _Float16* __restrict__ aoT) {     // [B,N,C]
  __shared__ __align__(16) _Float16 ldsK[32][HD];        // [m][c]   4 KB
  __shared__ __align__(16) _Float16 ldsV[HD][32];        // [c][m]   4 KB
  __shared__ __align__(16) _Float16 ldsP[8][16][32];     //          8 KB
  const int wid  = threadIdx.x >> 5;
  const int lane = threadIdx.x & 31;
  const int half = lane >> 4;
  const int gw   = blockIdx.x * 8 + wid;
  const int b  = gw / (HEADS * (NDIM / 16));
  const int re = gw % (HEADS * (NDIM / 16));
  const int h  = re >> 6;
  const int n0 = (re & 63) << 4;

  const size_t bh = (size_t)b * HEADS + h;
  const _Float16* kT = wk + bh * NDIM * HD;   // [n][c]
  const _Float16* vB = wv + bh * HD * NDIM;   // [c][n]

  const v16h aq0 = load_a_frag(wq + (bh * NDIM + n0) * HD, HD);      // c 0..31
  const v16h aq1 = load_a_frag(wq + (bh * NDIM + n0) * HD + 32, HD); // c 32..63

  float mrun[8], lrun[8];
  v8f O[4] = {};
#pragma unroll
  for (int r = 0; r < 8; ++r) { mrun[r] = -1e30f; lrun[r] = 0.f; }

  const float scale = 0.125f;  // 64^-0.5
  const int t = threadIdx.x;
  const int km = t >> 3, ks = t & 7;          // K panel: 32 rows x 8 segs
  const int vc = t >> 2, vs = t & 3;          // V panel: 64 rows x 4 segs

  for (int mo = 0; mo < NDIM; mo += 32) {
    // ---- cooperative stage of K/V panels (16B per lane) ----
#ifdef HAVE_ASYNC_LDS
    async_copy_b128(kT + (size_t)(mo + km) * HD + ks * 8, &ldsK[km][ks * 8]);
    async_copy_b128(vB + (size_t)vc * NDIM + mo + vs * 8, &ldsV[vc][vs * 8]);
    asm volatile("s_wait_asynccnt 0" ::: "memory");
#else
    *(v8h*)&ldsK[km][ks * 8] =
        *(const v8h*)(kT + (size_t)(mo + km) * HD + ks * 8);
    *(v8h*)&ldsV[vc][vs * 8] =
        *(const v8h*)(vB + (size_t)vc * NDIM + mo + vs * 8);
#endif
    if (mo + 32 < NDIM) __builtin_prefetch(kT + (size_t)(mo + 32 + km) * HD, 0, 1);
    __syncthreads();

    // ---- S = q^T k  (scores for 16 queries x 32 keys) ----
    v8f S0 = {}, S1 = {};
    S0 = wmma_f16(aq0, load_b_frag_t(&ldsK[0][0],  HD), S0);
    S0 = wmma_f16(aq1, load_b_frag_t(&ldsK[0][32], HD), S0);
    S1 = wmma_f16(aq0, load_b_frag_t(&ldsK[16][0],  HD), S1);
    S1 = wmma_f16(aq1, load_b_frag_t(&ldsK[16][32], HD), S1);

    // ---- online softmax ----
    float P0[8], P1[8];
#pragma unroll
    for (int r = 0; r < 8; ++r) {
      float s0 = S0[r] * scale, s1 = S1[r] * scale;
      float rm = rmax16(fmaxf(s0, s1));
      float mnew = fmaxf(mrun[r], rm);
      float sc = __expf(mrun[r] - mnew);
      P0[r] = __expf(s0 - mnew);
      P1[r] = __expf(s1 - mnew);
      lrun[r] = lrun[r] * sc + rsum16(P0[r] + P1[r]);
      mrun[r] = mnew;
#pragma unroll
      for (int tt = 0; tt < 4; ++tt) O[tt][r] *= sc;
    }

    // ---- P through LDS into A-fragment layout ----
#pragma unroll
    for (int r = 0; r < 8; ++r) {
      const int row = r + half * 8;
      const int col = lane & 15;
      ldsP[wid][row][col]      = (_Float16)P0[r];
      ldsP[wid][row][col + 16] = (_Float16)P1[r];
    }
    asm volatile("s_wait_dscnt 0" ::: "memory");
    const v16h ap = load_a_frag(&ldsP[wid][0][0], 32);

    // ---- O += P * v^T ----
#pragma unroll
    for (int tt = 0; tt < 4; ++tt)
      O[tt] = wmma_f16(ap, load_b_frag_t(&ldsV[tt * 16][0], 32), O[tt]);

    __syncthreads();   // protect ldsK/ldsV for next panel
  }

  // normalize + write out^T [B,N,C]
#pragma unroll
  for (int r = 0; r < 8; ++r) {
    const float inv = 1.0f / lrun[r];
    const int n = n0 + r + half * 8;
#pragma unroll
    for (int tt = 0; tt < 4; ++tt) {
      const int c = h * HD + tt * 16 + (lane & 15);
      aoT[((size_t)b * NDIM + n) * CDIM + c] = (_Float16)(O[tt][r] * inv);
    }
  }
}

// ---------------- Kernel 4: proj GEMM + bias + residual ----------------
__global__ void k_proj(const _Float16* __restrict__ aoT,  // [B,N,C]
                       const _Float16* __restrict__ whp,  // [C,C] f16
                       const float* __restrict__ proj_b,  // [C]
                       const float* __restrict__ x,       // [B,C,N]
                       float* __restrict__ out) {
  const int wid  = threadIdx.x >> 5;
  const int lane = threadIdx.x & 31;
  const int gw   = blockIdx.x * 8 + wid;
  const int wavesPerBatch = (CDIM / 16) * (NDIM / 64);    // 32*16
  const int b  = gw / wavesPerBatch;
  const int r  = gw % wavesPerBatch;
  const int o0 = (r / 16) * 16;
  const int n0 = (r % 16) * 64;

  const _Float16* ab = aoT + ((size_t)b * NDIM + n0) * CDIM;
  v8f acc[4] = {};
  for (int kk = 0; kk < CDIM; kk += 32) {
    const v16h a = load_a_frag(whp + (size_t)o0 * CDIM + kk, CDIM);
#pragma unroll
    for (int t = 0; t < 4; ++t) {
      const v16h bm = load_b_frag_t(ab + (size_t)(t * 16) * CDIM + kk, CDIM);
      acc[t] = wmma_f16(a, bm, acc[t]);
    }
  }

  const int half = lane >> 4;
#pragma unroll
  for (int t = 0; t < 4; ++t) {
    const int n = n0 + t * 16 + (lane & 15);
#pragma unroll
    for (int rr = 0; rr < 8; ++rr) {
      const int o = o0 + rr + half * 8;
      const size_t idx = ((size_t)b * CDIM + o) * NDIM + n;
      out[idx] = acc[t][rr] + proj_b[o] + x[idx];
    }
  }
}

// ---------------- launch ----------------
extern "C" void kernel_launch(void* const* d_in, const int* in_sizes, int n_in,
                              void* d_out, int out_size, void* d_ws, size_t ws_size,
                              hipStream_t stream) {
  const float* x      = (const float*)d_in[0];
  const float* gamma  = (const float*)d_in[1];
  const float* beta   = (const float*)d_in[2];
  const float* qkv_w  = (const float*)d_in[3];
  const float* qkv_b  = (const float*)d_in[4];
  const float* proj_w = (const float*)d_in[5];
  const float* proj_b = (const float*)d_in[6];
  float* out = (float*)d_out;

  _Float16* wsh = (_Float16*)d_ws;
  _Float16* xnT = wsh;              // [B,N,C]
  _Float16* wq  = xnT + PLANE;      // [B,H,N,HD]
  _Float16* wk  = wq + PLANE;       // [B,H,N,HD]
  _Float16* wv  = wk + PLANE;       // [B,H,HD,N]
  _Float16* ao  = wv + PLANE;       // [B,N,C]
  _Float16* whq = ao + PLANE;       // [3C,C]
  _Float16* whp = whq + QKV_W_ELEMS;// [C,C]

  k_cvt<<<(QKV_W_ELEMS + 255) / 256, 256, 0, stream>>>(qkv_w, proj_w, whq, whp);

  k_groupnorm<<<BATCH * GROUPS, 256, 0, stream>>>(x, gamma, beta, xnT);

  const int qkvWaves = BATCH * (3 * CDIM / 16) * (NDIM / 64);
  k_qkv<<<qkvWaves / 8, 256, 0, stream>>>(xnT, whq, qkv_b, wq, wk, wv);

  const int attnWaves = BATCH * HEADS * (NDIM / 16);
  k_attn<<<attnWaves / 8, 256, 0, stream>>>(wq, wk, wv, ao);

  const int projWaves = BATCH * (CDIM / 16) * (NDIM / 64);
  k_proj<<<projWaves / 8, 256, 0, stream>>>(ao, whp, proj_b, x, out);
}
